// MultiHeadAttention_12395275616621
// MI455X (gfx1250) — compile-verified, matrix operand order checked by reading, not run on hardware
//
#include <hip/hip_runtime.h>
#include <hip/hip_bf16.h>
#include <cstdint>

typedef __attribute__((ext_vector_type(16))) __bf16 v16bf;
typedef __attribute__((ext_vector_type(8)))  __bf16 v8bf;
typedef __attribute__((ext_vector_type(8)))  float  v8f;
typedef __attribute__((ext_vector_type(4)))  unsigned int v4u;
typedef __attribute__((ext_vector_type(8)))  int v8i;
typedef __attribute__((ext_vector_type(4)))  int v4i;

__device__ __forceinline__ v8f wmma_bf16(v16bf a, v16bf b, v8f c) {
  return __builtin_amdgcn_wmma_f32_16x16x32_bf16(false, a, false, b, (short)0, c, false, false);
}

// Build a 16-element bf16 fragment from two contiguous 16-byte runs.
// For half-wave hf: elements 0..7 = K = hf*8 .. hf*8+7, elements 8..15 =
// K = 16+hf*8 .. 16+hf*8+7 (matches the ISA 16-bit A/B 16x32 layout).
__device__ __forceinline__ v16bf frag16(const __bf16* base, int hf) {
  v8bf lo = *(const v8bf*)(base + hf * 8);
  v8bf hi = *(const v8bf*)(base + 16 + hf * 8);
  return __builtin_shufflevector(lo, hi, 0, 1, 2, 3, 4, 5, 6, 7,
                                          8, 9, 10, 11, 12, 13, 14, 15);
}

// ---------------------------------------------------------------------------
// TDM: issue a 2D tile DMA (global bf16, row-major) into LDS.
// D# built per CDNA5 ISA ch.8: group0 = {count, lds_addr, global_addr, type=2},
// group1 = {data_size=1 (2B), tensor_dim0/1, tile_dim0/1, dim0_stride}.
// This toolchain's builtin takes 6 args (g0, g1, g2, g3, extra v8i, cpol).
// ---------------------------------------------------------------------------
__device__ __forceinline__ void tdm_load_2d(unsigned lds_addr, const void* gaddr,
                                            unsigned tile_k, unsigned tile_rows,
                                            unsigned row_stride_elems) {
  unsigned long long ga = (unsigned long long)(uintptr_t)gaddr;
  v4u g0;
  g0[0] = 1u;                                          // count=1 (valid, user mode)
  g0[1] = lds_addr;                                    // LDS byte address
  g0[2] = (unsigned)(ga & 0xffffffffu);                // global_addr[31:0]
  g0[3] = (unsigned)((ga >> 32) & 0x01ffffffu) | (2u << 30);  // addr[56:32] | type=2

  const unsigned td0 = row_stride_elems;               // tensor_dim0 (elems)
  const unsigned td1 = 0x100000u;                      // tensor_dim1 (tiles in-bounds)
  const unsigned long long st0 = row_stride_elems;     // tensor_dim0_stride
  v8i g1;
  g1[0] = (int)(1u << 16);                             // wg_mask=0 | data_size=1(2B)
  g1[1] = (int)((td0 & 0xffffu) << 16);                // bar_addr=0 | dim0[15:0]
  g1[2] = (int)((td0 >> 16) | ((td1 & 0xffffu) << 16));// dim0[31:16] | dim1[15:0]
  g1[3] = (int)((td1 >> 16) | (tile_k << 16));         // dim1[31:16] | tile_dim0
  g1[4] = (int)(tile_rows & 0xffffu);                  // tile_dim1 | tile_dim2=0
  g1[5] = (int)(st0 & 0xffffffffu);                    // dim0_stride[31:0]
  g1[6] = (int)((st0 >> 32) & 0xffffu);                // dim0_stride[47:32] | dim1_stride lo
  g1[7] = 0;
  v4i g2 = {0, 0, 0, 0};
  v4i g3 = {0, 0, 0, 0};
  v8i g4 = {0, 0, 0, 0, 0, 0, 0, 0};
  __builtin_amdgcn_tensor_load_to_lds(g0, g1, g2, g3, g4, 0);
}

// ---------------------------------------------------------------------------
// Kernel 0a: fp32 -> bf16 streaming convert of queries/keys/values (z picks).
// ---------------------------------------------------------------------------
__global__ void __launch_bounds__(256) cvt_kernel(
    const float* __restrict__ x0, const float* __restrict__ x1,
    const float* __restrict__ x2, __bf16* __restrict__ y) {
  const int z = blockIdx.z;
  const float* x = (z == 0) ? x0 : ((z == 1) ? x1 : x2);
  __bf16* dst = y + (size_t)z * (8u << 20);
  size_t base = ((size_t)blockIdx.x * 256 + threadIdx.x) * 8;
  const float4* x4 = (const float4*)(x + base);
  float4 a = x4[0], b = x4[1];
  v8bf o;
  o[0] = (__bf16)a.x; o[1] = (__bf16)a.y; o[2] = (__bf16)a.z; o[3] = (__bf16)a.w;
  o[4] = (__bf16)b.x; o[5] = (__bf16)b.y; o[6] = (__bf16)b.z; o[7] = (__bf16)b.w;
  *(v8bf*)(dst + base) = o;
}

// ---------------------------------------------------------------------------
// Kernel 0b: W[k][n] fp32 -> Wt[n][k] bf16 (32x32 LDS tile transpose).
// ---------------------------------------------------------------------------
__global__ void __launch_bounds__(256) wtrans_kernel(
    const float* __restrict__ Wq, const float* __restrict__ Wk,
    const float* __restrict__ Wv, __bf16* __restrict__ Wt) {
  const int z = blockIdx.z;
  const float* W = (z == 0) ? Wq : ((z == 1) ? Wk : Wv);
  __bf16* dst = Wt + (size_t)z * (1u << 20);
  __shared__ float t[32][33];
  const int k0 = blockIdx.x * 32, n0 = blockIdx.y * 32;
  const int tx = threadIdx.x & 31, ty = threadIdx.x >> 5;
  #pragma unroll
  for (int i = 0; i < 4; ++i) {
    int r = ty + i * 8;
    t[r][tx] = W[(size_t)(k0 + r) * 1024 + n0 + tx];
  }
  __syncthreads();
  #pragma unroll
  for (int i = 0; i < 4; ++i) {
    int r = ty + i * 8;
    dst[(size_t)(n0 + r) * 1024 + k0 + tx] = (__bf16)t[tx][r];
  }
}

// ---------------------------------------------------------------------------
// Kernel A: QKV projection out = relu(Xb @ Wt^T + b), all-bf16 tiles staged by
// the Tensor Data Mover with double buffering. Block tile 64(M) x 128(N),
// BK=32, 8 waves (4x2); per wave 4 WMMA accumulators.
//   z=0: Q -> [B,H,S,dh]   z=1: K -> [B,H,S,dh]   z=2: V -> [B,H,dh,S]
// ---------------------------------------------------------------------------
__global__ void __launch_bounds__(256) qkv_proj_kernel(
    const __bf16* __restrict__ Xb, const __bf16* __restrict__ Wt,
    const float* __restrict__ bq, const float* __restrict__ bk, const float* __restrict__ bv,
    __bf16* __restrict__ qb, __bf16* __restrict__ kb, __bf16* __restrict__ vb)
{
  const int z = blockIdx.z;
  const __bf16* X = Xb + (size_t)z * (8u << 20);   // [8192][1024] row-major
  const __bf16* W = Wt + (size_t)z * (1u << 20);   // [n][k] row-major
  const float* bias = (z == 0) ? bq : ((z == 1) ? bk : bv);

  const int m0 = blockIdx.x * 64;
  const int n0 = blockIdx.y * 128;
  const int tid  = threadIdx.x;
  const int lane = tid & 31, wid = tid >> 5;
  const int wm = wid >> 1, wn = wid & 1;
  const int hf = lane >> 4, lm = lane & 15;

  __shared__ __bf16 As[2][64 * 32];    // [m][k]
  __shared__ __bf16 Bs[2][128 * 32];   // [n][k]

  v8f acc[4] = {};

  if (wid == 0) {   // pre-issue tile 0
    tdm_load_2d((unsigned)(uintptr_t)&As[0][0], &X[(size_t)m0 * 1024], 32, 64, 1024);
    tdm_load_2d((unsigned)(uintptr_t)&Bs[0][0], &W[(size_t)n0 * 1024], 32, 128, 1024);
  }

  for (int t = 0; t < 32; ++t) {
    const int cur = t & 1;
    if (wid == 0) {
      if (t + 1 < 32) {                // stream tile t+1 while computing t
        int k1 = (t + 1) * 32;
        tdm_load_2d((unsigned)(uintptr_t)&As[cur ^ 1][0], &X[(size_t)m0 * 1024 + k1], 32, 64, 1024);
        tdm_load_2d((unsigned)(uintptr_t)&Bs[cur ^ 1][0], &W[(size_t)n0 * 1024 + k1], 32, 128, 1024);
        __builtin_amdgcn_s_wait_tensorcnt(2);   // tile t complete, t+1 in flight
      } else {
        __builtin_amdgcn_s_wait_tensorcnt(0);
      }
    }
    __syncthreads();

    v16bf af = frag16(&As[cur][(wm * 16 + lm) * 32], hf);
    #pragma unroll
    for (int nt = 0; nt < 4; ++nt) {
      int n = wn * 64 + nt * 16 + lm;
      v16bf bfc = frag16(&Bs[cur][n * 32], hf);
      acc[nt] = wmma_bf16(af, bfc, acc[nt]);
    }
    __syncthreads();
  }

  // Epilogue: bias + ReLU, scatter into head-major bf16 layouts.
  #pragma unroll
  for (int nt = 0; nt < 4; ++nt) {
    #pragma unroll
    for (int r = 0; r < 8; ++r) {
      int m  = r + hf * 8;
      int gm = m0 + wm * 16 + m;
      int b  = gm >> 11, s = gm & 2047;      // S = 2048
      int e  = n0 + wn * 64 + nt * 16 + lm;
      float val = acc[nt][r] + bias[e];
      val = fmaxf(val, 0.f);
      __bf16 o = (__bf16)val;
      int h = e >> 6, d = e & 63;            // dh = 64
      if (z == 0)      qb[(((size_t)(b * 16 + h)) * 2048 + s) * 64 + d] = o;
      else if (z == 1) kb[(((size_t)(b * 16 + h)) * 2048 + s) * 64 + d] = o;
      else             vb[(((size_t)(b * 16 + h)) * 64 + d) * 2048 + s] = o;
    }
  }
}

// ---------------------------------------------------------------------------
// Kernel B: flash attention. Block = (b, h, 16-query tile), 4 waves; wave w
// owns keys {(4t+w)*32}, private online-softmax state + 16x64 O accumulator;
// split-K merge through LDS. All fragment loads are 16-byte vector loads.
// ---------------------------------------------------------------------------
__global__ void __launch_bounds__(128) attn_kernel(
    const __bf16* __restrict__ qb, const __bf16* __restrict__ kb,
    const __bf16* __restrict__ vb, const int* __restrict__ amask,
    float* __restrict__ att)
{
  const int q0 = blockIdx.x * 16;
  const int h  = blockIdx.y;
  const int b  = blockIdx.z;
  const int bh = b * 16 + h;
  const int tid = threadIdx.x, lane = tid & 31, wid = tid >> 5;
  const int hf = lane >> 4, lm = lane & 15;

  __shared__ __bf16 lds_p[4 * 16 * 32];   // per-wave P bounce (C-layout -> A-layout)
  __shared__ float  lds_o[4 * 16 * 64];   // per-wave O partials
  __shared__ float  lds_m[64];
  __shared__ float  lds_l[64];

  v16bf aq[2];
  #pragma unroll
  for (int p = 0; p < 2; ++p)
    aq[p] = frag16(&qb[((size_t)bh * 2048 + q0 + lm) * 64 + p * 32], hf);

  bool rowmask[8];
  #pragma unroll
  for (int r = 0; r < 8; ++r)
    rowmask[r] = amask[b * 2048 + q0 + r + hf * 8] != 0;

  float mrow[8], lrow[8];
  #pragma unroll
  for (int r = 0; r < 8; ++r) { mrow[r] = -3.0e38f; lrow[r] = 0.f; }
  v8f oacc[4] = {};

  for (int it = 0; it < 16; ++it) {
    const int kt = (it * 4 + wid) * 32;

    if (it + 1 < 16) {                      // prefetch next key/value tiles
      int ktn = ((it + 1) * 4 + wid) * 32;
      __builtin_prefetch(&kb[((size_t)bh * 2048 + ktn + lane) * 64], 0, 3);
      __builtin_prefetch(&vb[((size_t)bh * 64 + lane) * 2048 + ktn], 0, 3);
      __builtin_prefetch(&vb[((size_t)bh * 64 + 32 + lane) * 2048 + ktn], 0, 3);
    }

    // ---- scores: two 16x16 key tiles, K-dim = dh = 64 (2 wmma each) ----
    v8f sc[2];
    float tmax[8];
    #pragma unroll
    for (int r = 0; r < 8; ++r) tmax[r] = -3.0e38f;
    #pragma unroll
    for (int j = 0; j < 2; ++j) {
      v8f c = {};
      #pragma unroll
      for (int p = 0; p < 2; ++p) {
        v16bf bkf = frag16(&kb[((size_t)bh * 2048 + kt + j * 16 + lm) * 64 + p * 32], hf);
        c = wmma_bf16(aq[p], bkf, c);
      }
      #pragma unroll
      for (int r = 0; r < 8; ++r) {
        float sv = c[r] * 0.125f;               // 1/sqrt(64)
        sv = rowmask[r] ? sv : 1e-9f;           // masked rows -> uniform softmax
        c[r] = sv;
        tmax[r] = fmaxf(tmax[r], sv);
      }
      sc[j] = c;
    }

    // ---- online softmax (row reductions within 16-lane half groups) ----
    float alpha[8], pj[2][8];
    #pragma unroll
    for (int r = 0; r < 8; ++r) {
      float tm = tmax[r];
      #pragma unroll
      for (int off = 8; off >= 1; off >>= 1) tm = fmaxf(tm, __shfl_xor(tm, off, 32));
      float mnew = fmaxf(mrow[r], tm);
      alpha[r] = __expf(mrow[r] - mnew);
      mrow[r] = mnew;
      float rs = 0.f;
      #pragma unroll
      for (int j = 0; j < 2; ++j) {
        float p_ = __expf(sc[j][r] - mnew);
        pj[j][r] = p_;
        rs += p_;
      }
      #pragma unroll
      for (int off = 8; off >= 1; off >>= 1) rs += __shfl_xor(rs, off, 32);
      lrow[r] = lrow[r] * alpha[r] + rs;
    }
    #pragma unroll
    for (int nt = 0; nt < 4; ++nt)
      #pragma unroll
      for (int r = 0; r < 8; ++r) oacc[nt][r] *= alpha[r];

    // ---- P: f32 C-layout -> bf16 A-layout via per-wave LDS bounce ----
    #pragma unroll
    for (int j = 0; j < 2; ++j)
      #pragma unroll
      for (int r = 0; r < 8; ++r)
        lds_p[wid * 512 + (r + hf * 8) * 32 + j * 16 + lm] = (__bf16)pj[j][r];
    __syncthreads();
    v16bf pf = frag16(&lds_p[wid * 512 + lm * 32], hf);
    __syncthreads();

    // ---- P @ V: 4 dh tiles of 16, K-dim = 32 keys (V stored [B,H,dh,S]) ----
    #pragma unroll
    for (int nt = 0; nt < 4; ++nt) {
      v16bf bvf = frag16(&vb[((size_t)bh * 64 + nt * 16 + lm) * 2048 + kt], hf);
      oacc[nt] = wmma_bf16(pf, bvf, oacc[nt]);
    }
  }

  // ---- split-K merge of the 4 waves ----
  #pragma unroll
  for (int nt = 0; nt < 4; ++nt)
    #pragma unroll
    for (int r = 0; r < 8; ++r)
      lds_o[(wid * 16 + r + hf * 8) * 64 + nt * 16 + lm] = oacc[nt][r];
  if (lm == 0) {
    #pragma unroll
    for (int r = 0; r < 8; ++r) {
      lds_m[wid * 16 + r + hf * 8] = mrow[r];
      lds_l[wid * 16 + r + hf * 8] = lrow[r];
    }
  }
  __syncthreads();

  int row = tid >> 3;
  int dbase = (tid & 7) * 8;
  float M = -3.0e38f;
  #pragma unroll
  for (int w = 0; w < 4; ++w) M = fmaxf(M, lds_m[w * 16 + row]);
  float ew[4], L = 0.f;
  #pragma unroll
  for (int w = 0; w < 4; ++w) {
    ew[w] = __expf(lds_m[w * 16 + row] - M);
    L += lds_l[w * 16 + row] * ew[w];
  }
  float inv = 1.f / L;
  #pragma unroll
  for (int dd = 0; dd < 8; ++dd) {
    int d = dbase + dd;
    float o = 0.f;
    #pragma unroll
    for (int w = 0; w < 4; ++w) o += lds_o[(w * 16 + row) * 64 + d] * ew[w];
    att[((size_t)(b * 2048 + q0 + row)) * 1024 + h * 64 + d] = o * inv;
  }
}

// ---------------------------------------------------------------------------
// Kernel C: residual + LayerNorm over D=1024 per row.
// ---------------------------------------------------------------------------
__global__ void __launch_bounds__(256) ln_kernel(
    const float* __restrict__ att, const float* __restrict__ queries,
    const float* __restrict__ gamma, const float* __restrict__ beta,
    float* __restrict__ out)
{
  const int row = blockIdx.x;
  const int tid = threadIdx.x, lane = tid & 31, wid = tid >> 5;
  float x[4], s = 0.f, s2 = 0.f;
  #pragma unroll
  for (int i = 0; i < 4; ++i) {
    int c = tid + i * 256;
    float v = att[(size_t)row * 1024 + c] + queries[(size_t)row * 1024 + c];
    x[i] = v; s += v; s2 += v * v;
  }
  #pragma unroll
  for (int off = 16; off >= 1; off >>= 1) {
    s  += __shfl_xor(s,  off, 32);
    s2 += __shfl_xor(s2, off, 32);
  }
  __shared__ float sh1[8], sh2[8];
  if (lane == 0) { sh1[wid] = s; sh2[wid] = s2; }
  __syncthreads();
  float S = 0.f, S2 = 0.f;
  #pragma unroll
  for (int w = 0; w < 8; ++w) { S += sh1[w]; S2 += sh2[w]; }
  float mu   = S * (1.f / 1024.f);
  float var  = S2 * (1.f / 1024.f) - mu * mu;
  float rstd = rsqrtf(var + 1e-5f);
  #pragma unroll
  for (int i = 0; i < 4; ++i) {
    int c = tid + i * 256;
    out[(size_t)row * 1024 + c] = (x[i] - mu) * rstd * gamma[c] + beta[c];
  }
}

extern "C" void kernel_launch(void* const* d_in, const int* in_sizes, int n_in,
                              void* d_out, int out_size, void* d_ws, size_t ws_size,
                              hipStream_t stream) {
  const float* queries = (const float*)d_in[0];
  const float* keys    = (const float*)d_in[1];
  const float* values  = (const float*)d_in[2];
  const int*   amask   = (const int*)d_in[3];
  const float* Wq = (const float*)d_in[4];
  const float* bq = (const float*)d_in[5];
  const float* Wk = (const float*)d_in[6];
  const float* bk = (const float*)d_in[7];
  const float* Wv = (const float*)d_in[8];
  const float* bv = (const float*)d_in[9];
  const float* gamma = (const float*)d_in[10];
  const float* beta  = (const float*)d_in[11];
  float* out = (float*)d_out;

  char* ws = (char*)d_ws;
  // [0, 48MB)  : Xb bf16 (3 x 8M elems); after the GEMM this region is dead,
  //              so att (32MB f32) overlays it.
  // [48, 54MB) : Wt bf16 (3 x 1M elems, transposed weights)
  // [54,102MB) : qb / kb / vb bf16 (16MB each)
  __bf16* Xb = (__bf16*)(ws);
  __bf16* Wt = (__bf16*)(ws + (size_t)48 * 1024 * 1024);
  __bf16* qb = (__bf16*)(ws + (size_t)54 * 1024 * 1024);
  __bf16* kb = (__bf16*)(ws + (size_t)70 * 1024 * 1024);
  __bf16* vb = (__bf16*)(ws + (size_t)86 * 1024 * 1024);
  float*  att = (float*)(ws);                      // overlays Xb (disjoint lifetime)

  cvt_kernel<<<dim3(4096, 1, 3), dim3(256), 0, stream>>>(queries, keys, values, Xb);
  wtrans_kernel<<<dim3(32, 32, 3), dim3(256), 0, stream>>>(Wq, Wk, Wv, Wt);

  dim3 gA(128, 8, 3);                    // M/64, N/128, {q,k,v}
  qkv_proj_kernel<<<gA, dim3(256), 0, stream>>>(Xb, Wt, bq, bk, bv, qb, kb, vb);

  dim3 gB(128, 16, 4);                   // S/16 query tiles, H, B
  attn_kernel<<<gB, dim3(128), 0, stream>>>(qb, kb, vb, amask, att);

  ln_kernel<<<dim3(8192), dim3(256), 0, stream>>>(att, queries, gamma, beta, out);
}
